// RademacherProjection_17317308137680
// MI455X (gfx1250) — compile-verified, matrix operand order checked by reading, not run on hardware
//
#include <hip/hip_runtime.h>
#include <stdint.h>

// ---------------------------------------------------------------------------
// Blocked Rademacher random projection for MI455X (gfx1250, wave32, WMMA).
//   out[16][256] = sum_d g[16][d] * sign(d,p) * 256^-0.5
// Threefry2x32-generated signs (per-block folded key, mirrors fold_in).
// g split into bf16 hi+lo; two v_wmma_f32_16x16x32_bf16 per K=32 step give
// ~f32 accuracy at bf16 matrix-core rate. Double-buffered LDS pipeline hides
// global-load latency behind the WMMA+PRNG phase.
// ---------------------------------------------------------------------------

#define PARAM_DIM     4194304
#define PROJ_DIM      256
#define BATCH         16
#define BLOCK_D       4096                     // reference per-seed block
#define N_BLOCKS      (PARAM_DIM / BLOCK_D)    // 1024
#define GRID1         256                      // stage-1 workgroups
#define BLK_PER_WG    (N_BLOCKS / GRID1)       // 4
#define KSTEP         512                      // f32 of g staged per tile
#define TILES_PER_BLK (BLOCK_D / KSTEP)        // 8
#define NTILES        (BLK_PER_WG * TILES_PER_BLK)  // 32 tiles per WG
#define LDS_STRIDE    260                      // u32/row: 256 data + 4 skew
#define SEED_LO       42u

typedef __attribute__((ext_vector_type(16))) __bf16   v16bf;
typedef __attribute__((ext_vector_type(8)))  float    v8f;
typedef __attribute__((ext_vector_type(8)))  uint32_t v8u;
typedef __attribute__((ext_vector_type(4)))  uint32_t u32x4;
typedef __attribute__((ext_vector_type(4)))  float    f32x4;

__device__ __forceinline__ uint16_t f32_to_bf16(float f) {
    uint32_t u = __builtin_bit_cast(uint32_t, f);
    uint32_t r = u + 0x7FFFu + ((u >> 16) & 1u);   // round-to-nearest-even
    return (uint16_t)(r >> 16);
}
__device__ __forceinline__ float bf16_to_f32(uint16_t h) {
    return __builtin_bit_cast(float, (uint32_t)h << 16);
}

// Threefry-2x32, 20 rounds (JAX PRNG core). Rotates lower to v_alignbit.
__device__ __forceinline__ void tf2x32(uint32_t k0, uint32_t k1,
                                       uint32_t c0, uint32_t c1,
                                       uint32_t& o0, uint32_t& o1) {
    const uint32_t ks[3] = { k0, k1, k0 ^ k1 ^ 0x1BD11BDAu };
    const uint32_t R[8]  = { 13, 15, 26, 6, 17, 29, 16, 24 };
    uint32_t x0 = c0 + k0, x1 = c1 + k1;
#pragma unroll
    for (int chunk = 0; chunk < 5; ++chunk) {
#pragma unroll
        for (int r = 0; r < 4; ++r) {
            uint32_t rot = R[(chunk & 1) * 4 + r];
            x0 += x1;
            x1 = (x1 << rot) | (x1 >> (32u - rot));
            x1 ^= x0;
        }
        x0 += ks[(chunk + 1) % 3];
        x1 += ks[(chunk + 2) % 3] + (uint32_t)(chunk + 1);
    }
    o0 = x0; o1 = x1;
}

// Convert a prefetched 4xf32x4 row segment to packed hi/lo bf16 in LDS.
__device__ __forceinline__ void stageTile(const f32x4* pre, int lane, int row,
                                          uint32_t (*dHi)[LDS_STRIDE],
                                          uint32_t (*dLo)[LDS_STRIDE]) {
#pragma unroll
    for (int q = 0; q < 4; ++q) {
        const f32x4 v = pre[q];
        const uint16_t h0 = f32_to_bf16(v.x), h1 = f32_to_bf16(v.y);
        const uint16_t h2 = f32_to_bf16(v.z), h3 = f32_to_bf16(v.w);
        const uint16_t l0 = f32_to_bf16(v.x - bf16_to_f32(h0));
        const uint16_t l1 = f32_to_bf16(v.y - bf16_to_f32(h1));
        const uint16_t l2 = f32_to_bf16(v.z - bf16_to_f32(h2));
        const uint16_t l3 = f32_to_bf16(v.w - bf16_to_f32(h3));
        const int w32 = (q * 128 + lane * 4) >> 1;
        dHi[row][w32]     = (uint32_t)h0 | ((uint32_t)h1 << 16);
        dHi[row][w32 + 1] = (uint32_t)h2 | ((uint32_t)h3 << 16);
        dLo[row][w32]     = (uint32_t)l0 | ((uint32_t)l1 << 16);
        dLo[row][w32 + 1] = (uint32_t)l2 | ((uint32_t)l3 << 16);
    }
}

// ---------------------------------------------------------------------------
// Stage 1: 256 WGs x 512 threads (16 waves). Wave w owns proj cols [16w,16w+16).
// Double-buffered LDS staging of g (hi/lo bf16), shared by all 16 waves, so g
// is read from HBM exactly once. Each WG reduces 4 param-blocks into a private
// 16x256 f32 partial (deterministic; no atomics).
// ---------------------------------------------------------------------------
__global__ __launch_bounds__(512)
void rademacher_stage1(const float* __restrict__ g, float* __restrict__ part) {
    __shared__ __align__(16) uint32_t sHi[2][BATCH][LDS_STRIDE];
    __shared__ __align__(16) uint32_t sLo[2][BATCH][LDS_STRIDE];

    const int tid   = threadIdx.x;
    const int wave  = tid >> 5;          // 0..15 -> proj tile / staged row
    const int lane  = tid & 31;
    const int n     = lane & 15;         // N within tile; also A-row M
    const int khalf = lane >> 4;         // 0: low K-half, 1: high K-half
    const int p     = wave * 16 + n;     // global proj column

    const size_t dWg  = (size_t)blockIdx.x * (BLK_PER_WG * BLOCK_D);
    const float* gRow = g + (size_t)wave * PARAM_DIM + dWg + (size_t)(lane * 4);

    v8f   acc = {};                      // C accumulator (16x16 f32 tile)
    f32x4 pre[4];

    // --- prologue: prefetch + stage tile 0 into buffer 0 ---
#pragma unroll
    for (int q = 0; q < 4; ++q)
        pre[q] = *(const f32x4*)(gRow + q * 128);
    stageTile(pre, lane, wave, sHi[0], sLo[0]);

    uint32_t fk0 = 0, fk1 = 0;

    for (int t = 0; t < NTILES; ++t) {
        if ((t & (TILES_PER_BLK - 1)) == 0) {   // new param-block: fold key
            tf2x32(0u, SEED_LO, 0u,
                   (uint32_t)(blockIdx.x * BLK_PER_WG) + (uint32_t)(t >> 3),
                   fk0, fk1);
        }
        __syncthreads();   // buffer t&1 staged; all readers of (t+1)&1 done

        // prefetch tile t+1 (latency hidden behind WMMA+threefry below)
        if (t + 1 < NTILES) {
#pragma unroll
            for (int q = 0; q < 4; ++q)
                pre[q] = *(const f32x4*)(gRow + (size_t)(t + 1) * KSTEP + q * 128);
        }

        const uint32_t (*bHi)[LDS_STRIDE] = sHi[t & 1];
        const uint32_t (*bLo)[LDS_STRIDE] = sLo[t & 1];
        const uint32_t ktInBlk = (uint32_t)(t & (TILES_PER_BLK - 1)) * KSTEP;

        for (int kg = 0; kg < KSTEP; kg += 128) {
            // one threefry per lane covers this lane's B signs for 4 K-steps
            uint32_t r0, r1;
            tf2x32(fk0, fk1,
                   (uint32_t)p | ((uint32_t)khalf << 8)
                               | (((ktInBlk + (uint32_t)kg) >> 7) << 9),
                   0u, r0, r1);
#pragma unroll
            for (int s = 0; s < 4; ++s) {
                const int k0 = kg + s * 32;
                // A: lanes 0-15 hold K {0-7,16-23}, lanes 16-31 K {8-15,24-31}
                const int ka = (k0 + 8 * khalf) >> 1;       // u32 index
                const int kb = (k0 + 16 + 8 * khalf) >> 1;
                u32x4 h0 = *(const u32x4*)&bHi[n][ka];
                u32x4 h1 = *(const u32x4*)&bHi[n][kb];
                u32x4 l0 = *(const u32x4*)&bLo[n][ka];
                u32x4 l1 = *(const u32x4*)&bLo[n][kb];
                v8u ahi = { h0.x, h0.y, h0.z, h0.w, h1.x, h1.y, h1.z, h1.w };
                v8u alo = { l0.x, l0.y, l0.z, l0.w, l1.x, l1.y, l1.z, l1.w };

                // B signs: single shift drops bit v at [15] and v+16 at [31]
                // (step even -> bits {v, v+16}; step odd -> {v+8, v+24})
                const uint32_t w    = (s < 2) ? r0 : r1;
                const int      base = (s & 1) ? 7 : 15;
                v8u bm;
#pragma unroll
                for (int v = 0; v < 8; ++v)
                    bm[v] = ((w << (base - v)) & 0x80008000u) | 0x3F803F80u;

                v16bf A  = __builtin_bit_cast(v16bf, ahi);
                v16bf Al = __builtin_bit_cast(v16bf, alo);
                v16bf Bm = __builtin_bit_cast(v16bf, bm);
                acc = __builtin_amdgcn_wmma_f32_16x16x32_bf16(
                          false, A,  false, Bm, (short)0, acc, false, false);
                acc = __builtin_amdgcn_wmma_f32_16x16x32_bf16(
                          false, Al, false, Bm, (short)0, acc, false, false);
            }
        }

        // stage the prefetched tile t+1 into the other buffer
        if (t + 1 < NTILES)
            stageTile(pre, lane, wave, sHi[(t + 1) & 1], sLo[(t + 1) & 1]);
    }

    // C/D layout: VGPR j -> M = j + 8*khalf, N = lane&15 (global col p)
    float* dst = part + (size_t)blockIdx.x * (BATCH * PROJ_DIM);
#pragma unroll
    for (int j = 0; j < 8; ++j) {
        const int m = j + 8 * khalf;
        dst[m * PROJ_DIM + p] = acc[j];
    }
}

// ---------------------------------------------------------------------------
// Stage 2: deterministic fixed-order reduction of the 256 partials + scale.
// ---------------------------------------------------------------------------
__global__ __launch_bounds__(256)
void rademacher_stage2(const float* __restrict__ part, float* __restrict__ out) {
    const int j = blockIdx.x * blockDim.x + threadIdx.x;   // 0..4095
    float s = 0.0f;
    for (int w = 0; w < GRID1; ++w)
        s += part[(size_t)w * (BATCH * PROJ_DIM) + j];
    out[j] = s * 0.0625f;   // 256^-0.5
}

extern "C" void kernel_launch(void* const* d_in, const int* in_sizes, int n_in,
                              void* d_out, int out_size, void* d_ws, size_t ws_size,
                              hipStream_t stream) {
    const float* g   = (const float*)d_in[0];
    float*       out = (float*)d_out;
    float*       ws  = (float*)d_ws;   // 256 * 16 * 256 f32 = 4 MB partials

    rademacher_stage1<<<GRID1, 512, 0, stream>>>(g, ws);
    rademacher_stage2<<<(BATCH * PROJ_DIM) / 256, 256, 0, stream>>>(ws, out);
}
